// CrossViewAttention_53214644798134
// MI455X (gfx1250) — matrix operand order repaired and checked by reading, not compiled, not run on hardware
//
#include <hip/hip_runtime.h>

#define DD   1024
#define NB   16384
#define NTILES 16             // 1024 / 64  (N per workgroup = 64)
#define MTILES 128            // 32768 / 256 (M per workgroup = 8 waves * 32 rows)
#define KSTEPS (DD / 32)

#define LDS_BSTRIDE 40                   // padded LDS row stride (bf16): 32 data + 8 pad (80B)
#define LDS_MSIZE   (64 * LDS_BSTRIDE)   // one matrix's 64x32 tile
#define LDS_TILE    (3 * LDS_MSIZE)      // Q,K,V tiles

typedef __attribute__((ext_vector_type(16))) __bf16 v16bf;
typedef __attribute__((ext_vector_type(8)))  __bf16 v8bf;
typedef __attribute__((ext_vector_type(4)))  __bf16 v4bf;
typedef __attribute__((ext_vector_type(8)))  float  v8f;
typedef __attribute__((ext_vector_type(4)))  int    v4i;

typedef __attribute__((address_space(1))) v4i glob_v4i;   // global int4
typedef __attribute__((address_space(3))) v4i lds_v4i;    // LDS int4

#if __has_builtin(__builtin_amdgcn_global_load_async_to_lds_b128)
#define HAVE_ASYNC_LDS 1
#endif

// ---------------------------------------------------------------------------
// Fragment loaders (CDNA5 wave32 WMMA 16x16x32 bf16 register layouts)
// A (16x32, MxK): lane l<16 -> row l, K = {kc+0..7, kc+16..23}
//                 lane l>=16 -> row l-16, K = {kc+8..15, kc+24..31}
// caller passes p = rowbase + kc + (lane>>4)*8
__device__ __forceinline__ v16bf load_a_frag(const __bf16* p) {
    v8bf lo = *(const v8bf*)(p);
    v8bf hi = *(const v8bf*)(p + 16);
    return __builtin_shufflevector(lo, hi, 0,1,2,3,4,5,6,7,8,9,10,11,12,13,14,15);
}
// B (32x16, KxN): lane l<16 -> col l, K = {kc+0..7, kc+16..23} ; lane l>=16 -> col l-16, K = {kc+8..15, kc+24..31}
// caller passes p = colbase + (lane>>4)*8 (LDS)
__device__ __forceinline__ v16bf load_b_frag(const __bf16* p) {
    v8bf lo = *(const v8bf*)(p);
    v8bf hi = *(const v8bf*)(p + 16);
    return __builtin_shufflevector(lo, hi, 0,1,2,3,4,5,6,7,8,9,10,11,12,13,14,15);
}

// ---------------------------------------------------------------------------
// Cooperative copy of the 3x(64x32) bf16 weight tile for one K-step into LDS.
// 256 threads x 3 x 16B = 12 KB. tid>>2 = row (0..63), (tid&3)*8 = K sub-offset.
__device__ __forceinline__ void issue_btile(const __bf16* __restrict__ W,
                                            __bf16* dst, int tid, int n0, int kc) {
    const int row  = tid >> 2;
    const int koff = (tid & 3) * 8;
#pragma unroll
    for (int m = 0; m < 3; ++m) {
        const __bf16* g = W + (size_t)m * DD * DD + (size_t)(n0 + row) * DD + kc + koff;
        __bf16* l = dst + m * LDS_MSIZE + row * LDS_BSTRIDE + koff;
#ifdef HAVE_ASYNC_LDS
        __builtin_amdgcn_global_load_async_to_lds_b128(
            (glob_v4i*)g, (lds_v4i*)l, 0, 0);
#else
        *(v8bf*)l = *(const v8bf*)g;   // sync fallback: global_load + ds_store
#endif
    }
}

__device__ __forceinline__ void wait_async_copies() {
#ifdef HAVE_ASYNC_LDS
    asm volatile("s_wait_asynccnt 0x0" ::: "memory");
#endif
}

// ---------------------------------------------------------------------------
// f32 -> bf16 converters (pre-pass)
__global__ void __launch_bounds__(256)
cvt_f32_bf16_kernel(const float* __restrict__ src, __bf16* __restrict__ dst, int n4) {
    int i = blockIdx.x * blockDim.x + threadIdx.x;
    if (i >= n4) return;
    float4 f = ((const float4*)src)[i];
    v4bf o;
    o[0] = (__bf16)f.x; o[1] = (__bf16)f.y; o[2] = (__bf16)f.z; o[3] = (__bf16)f.w;
    *(v4bf*)(dst + (size_t)i * 4) = o;
}

// interleave f1/f2 rows into X[2B x D] bf16 (row 2b = f1[b], row 2b+1 = f2[b])
__global__ void __launch_bounds__(256)
interleave_kernel(const float* __restrict__ f1, const float* __restrict__ f2,
                  __bf16* __restrict__ X) {
    int i  = blockIdx.x * blockDim.x + threadIdx.x;   // over B*D/4
    int b  = i >> 8;                                  // D/4 == 256
    int d4 = i & 255;
    float4 a = ((const float4*)f1)[i];
    float4 c = ((const float4*)f2)[i];
    v4bf pa, pc;
    pa[0] = (__bf16)a.x; pa[1] = (__bf16)a.y; pa[2] = (__bf16)a.z; pa[3] = (__bf16)a.w;
    pc[0] = (__bf16)c.x; pc[1] = (__bf16)c.y; pc[2] = (__bf16)c.z; pc[3] = (__bf16)c.w;
    *(v4bf*)(X + ((size_t)(2 * b)    ) * DD + d4 * 4) = pa;
    *(v4bf*)(X + ((size_t)(2 * b) + 1) * DD + d4 * 4) = pc;
}

// ---------------------------------------------------------------------------
// Fused QKV GEMM. Each wave: 32 rows (2 A frags) x 64 output features for
// Q/K/V. Weight tiles are async-copied to LDS (double buffered); B fragments
// stream from LDS with a one-frag rotating pipeline; A fragments are
// register-prefetched one K-step ahead. Every B fragment feeds two WMMAs.
// V(+bias) -> global; Q/K collapse in-register into per-N-tile partial
// attention scores -> spart[ntile][b][4] (unique writers => deterministic).
__global__ void __launch_bounds__(256)
qkv_gemm_kernel(const __bf16* __restrict__ X, const __bf16* __restrict__ W,
                const float* __restrict__ bq, const float* __restrict__ bk,
                const float* __restrict__ bv,
                float* __restrict__ Vout, float* __restrict__ spart) {
    __shared__ __bf16 smem[2][LDS_TILE];   // 2 x 15 KB

    const int tid  = threadIdx.x;
    const int lane = tid & 31;
    const int wave = tid >> 5;
    const int l15  = lane & 15;
    const int hi16 = lane >> 4;

    const int n0 = blockIdx.x * 64;              // N-tile base (output feature)
    const int r0 = blockIdx.y * 256 + wave * 32; // 32-row block base (of 2B rows)

    const __bf16* aBase0 = X + (size_t)(r0 + l15)      * DD + hi16 * 8;
    const __bf16* aBase1 = X + (size_t)(r0 + 16 + l15) * DD + hi16 * 8;

    // acc[mb][matrix][nb] : mb = M-block (0..1), matrix = Q/K/V, nb = N-block
    v8f acc[2][3][4];
#pragma unroll
    for (int mb = 0; mb < 2; ++mb)
#pragma unroll
        for (int m = 0; m < 3; ++m)
#pragma unroll
            for (int nb = 0; nb < 4; ++nb) acc[mb][m][nb] = (v8f)0.0f;

    issue_btile(W, smem[0], tid, n0, 0);   // prologue: async-prefetch K-step 0
    v16bf a0 = load_a_frag(aBase0);        // prologue: A frags for K-step 0
    v16bf a1 = load_a_frag(aBase1);

    // B-fragment LDS offsets, iteration order t = nb*3 + m
    const __bf16* sb0 = (const __bf16*)smem + (size_t)l15 * LDS_BSTRIDE + hi16 * 8;

#pragma unroll 1
    for (int kt = 0; kt < KSTEPS; ++kt) {
        const int kc  = kt * 32;
        const int cur = kt & 1;

        wait_async_copies();       // my copies into smem[cur] have landed
        __syncthreads();           // everyone's landed; smem[cur^1] free to overwrite

        if (kt + 1 < KSTEPS)
            issue_btile(W, smem[cur ^ 1], tid, n0, kc + 32);  // async-prefetch next

        // register-prefetch next iteration's A frags (clamped on last step)
        const int kcn = (kt + 1 < KSTEPS) ? kc + 32 : kc;
        v16bf a0n = load_a_frag(aBase0 + kcn);
        v16bf a1n = load_a_frag(aBase1 + kcn);

        const __bf16* sb = sb0 + cur * LDS_TILE;
        // one-deep rotating pipeline over the 12 B fragments
        v16bf w = load_b_frag(sb);         // t = 0 (nb=0, m=0)
#pragma unroll
        for (int t = 0; t < 12; ++t) {
            const int nb = t >> 2 == 0 ? t / 3 : t / 3;   // t/3
            const int m  = t - (t / 3) * 3;               // t%3
            const int tn = (t + 1 < 12) ? t + 1 : 11;
            const int nbn = tn / 3, mn = tn - (tn / 3) * 3;
            v16bf wn = load_b_frag(sb + mn * LDS_MSIZE + nbn * 16 * LDS_BSTRIDE);
            acc[0][m][nb] = __builtin_amdgcn_wmma_f32_16x16x32_bf16(
                false, a0, false, w, (short)0, acc[0][m][nb], false, false);
            acc[1][m][nb] = __builtin_amdgcn_wmma_f32_16x16x32_bf16(
                false, a1, false, w, (short)0, acc[1][m][nb], false, false);
            w = wn;
        }
        a0 = a0n;
        a1 = a1n;
    }

    // Epilogue per M-block: bias add, V store, in-register partial scores.
    // C/D layout: lane l<16 : element j -> (M=j,    N=l)
    //             lane l>=16: element j -> (M=8+j,  N=l-16)
#pragma unroll
    for (int mb = 0; mb < 2; ++mb) {
        const int rb = r0 + mb * 16;
        float p[4][4];
#pragma unroll
        for (int i = 0; i < 4; ++i)
#pragma unroll
            for (int st = 0; st < 4; ++st) p[i][st] = 0.0f;

#pragma unroll
        for (int nb = 0; nb < 4; ++nb) {
            const int n = n0 + nb * 16 + l15;
            const float bQ = bq[n], bK = bk[n], bV = bv[n];
            v8f q = acc[mb][0][nb], k = acc[mb][1][nb], v = acc[mb][2][nb];
#pragma unroll
            for (int j = 0; j < 8; ++j) { q[j] += bQ; k[j] += bK; v[j] += bV; }
#pragma unroll
            for (int j = 0; j < 8; ++j) {
                const int row = rb + hi16 * 8 + j;
                Vout[(size_t)row * DD + n] = v[j];
            }
            // rows (2i, 2i+1) within this half's 8 rows form one batch element
#pragma unroll
            for (int i = 0; i < 4; ++i) {
                p[i][0] += q[2 * i]     * k[2 * i];
                p[i][1] += q[2 * i]     * k[2 * i + 1];
                p[i][2] += q[2 * i + 1] * k[2 * i];
                p[i][3] += q[2 * i + 1] * k[2 * i + 1];
            }
        }

        // reduce over the 16 e-lanes of each half (xor masks < 16 stay in-half)
#pragma unroll
        for (int i = 0; i < 4; ++i)
#pragma unroll
            for (int st = 0; st < 4; ++st) {
                float v = p[i][st];
                v += __shfl_xor(v, 1);
                v += __shfl_xor(v, 2);
                v += __shfl_xor(v, 4);
                v += __shfl_xor(v, 8);
                p[i][st] = v;
            }

        if (l15 == 0) {  // lanes 0 (half 0 -> batch +0..3) and 16 (half 1 -> +4..7)
            float* sp = spart + (size_t)blockIdx.x * (NB * 4);
            const int bbase = (rb >> 1) + hi16 * 4;
#pragma unroll
            for (int i = 0; i < 4; ++i)
#pragma unroll
                for (int st = 0; st < 4; ++st)
                    sp[(size_t)(bbase + i) * 4 + st] = p[i][st] * (1.0f / 32.0f);
        }
    }
}

// ---------------------------------------------------------------------------
// Attention epilogue: sum score partials, 2x2 softmax, weighted V sum.
__global__ void __launch_bounds__(256)
attend_kernel(const float* __restrict__ Vin, const float* __restrict__ spart,
              float* __restrict__ out) {
    const int b = blockIdx.x;
    float s0 = 0.f, s1 = 0.f, s2 = 0.f, s3 = 0.f;
#pragma unroll
    for (int nt = 0; nt < NTILES; ++nt) {
        const float* sp = spart + (size_t)nt * (NB * 4) + (size_t)b * 4;
        s0 += sp[0]; s1 += sp[1]; s2 += sp[2]; s3 += sp[3];
    }
    const float m0 = fmaxf(s0, s1);
    const float e00 = __expf(s0 - m0), e01 = __expf(s1 - m0);
    const float r0i = 1.0f / (e00 + e01);
    const float m1 = fmaxf(s2, s3);
    const float e10 = __expf(s2 - m1), e11 = __expf(s3 - m1);
    const float r1i = 1.0f / (e10 + e11);
    const float w0 = e00 * r0i + e10 * r1i;
    const float w1 = e01 * r0i + e11 * r1i;

    const int d = threadIdx.x * 4;
    const float4 v0 = *(const float4*)(Vin + (size_t)(2 * b)     * DD + d);
    const float4 v1 = *(const float4*)(Vin + (size_t)(2 * b + 1) * DD + d);
    float4 o;
    o.x = w0 * v0.x + w1 * v1.x;
    o.y = w0 * v0.y + w1 * v1.y;
    o.z = w0 * v0.z + w1 * v1.z;
    o.w = w0 * v0.w + w1 * v1.w;
    *(float4*)(out + (size_t)b * DD + d) = o;
}

// ---------------------------------------------------------------------------
extern "C" void kernel_launch(void* const* d_in, const int* in_sizes, int n_in,
                              void* d_out, int out_size, void* d_ws, size_t ws_size,
                              hipStream_t stream) {
    (void)in_sizes; (void)n_in; (void)out_size; (void)ws_size;
    const float* f1 = (const float*)d_in[0];
    const float* f2 = (const float*)d_in[1];
    const float* Wq = (const float*)d_in[2];
    const float* bq = (const float*)d_in[3];
    const float* Wk = (const float*)d_in[4];
    const float* bk = (const float*)d_in[5];
    const float* Wv = (const float*)d_in[6];
    const float* bv = (const float*)d_in[7];
    float* out = (float*)d_out;

    // workspace layout (bytes)
    char* ws = (char*)d_ws;
    __bf16* Xbf = (__bf16*)(ws);                       // 2B*D*2   = 64 MiB
    __bf16* Wbf = (__bf16*)(ws + (size_t)67108864);    // 3*D*D*2  =  6 MiB
    float*  Vf  = (float*) (ws + (size_t)73400320);    // 2B*D*4   = 128 MiB
    float*  sp  = (float*) (ws + (size_t)207618048);   // 16*B*4*4 =  4 MiB

    // 1) bf16 pre-pass
    interleave_kernel<<<(NB * DD / 4) / 256, 256, 0, stream>>>(f1, f2, Xbf);
    cvt_f32_bf16_kernel<<<(DD * DD / 4) / 256, 256, 0, stream>>>(Wq, Wbf,               DD * DD / 4);
    cvt_f32_bf16_kernel<<<(DD * DD / 4) / 256, 256, 0, stream>>>(Wk, Wbf + DD * DD,     DD * DD / 4);
    cvt_f32_bf16_kernel<<<(DD * DD / 4) / 256, 256, 0, stream>>>(Wv, Wbf + 2 * DD * DD, DD * DD / 4);

    // 2) fused QKV GEMM + partial scores + V store
    qkv_gemm_kernel<<<dim3(NTILES, MTILES), 256, 0, stream>>>(Xbf, Wbf, bq, bk, bv, Vf, sp);

    // 3) softmax + weighted V sum
    attend_kernel<<<NB, 256, 0, stream>>>(Vf, sp, out);
}
